// SentimentLSTM_31095563223878
// MI455X (gfx1250) — matrix operand order repaired and checked by reading, compiled, not run on hardware
//
#include <hip/hip_runtime.h>
#include <hip/hip_bf16.h>

typedef __attribute__((ext_vector_type(16))) _Float16 v16h;
typedef __attribute__((ext_vector_type(8)))  float    v8f;

// ---------------------------------------------------------------------------
// WMMA fragment loaders (fp16, 16x16x32, wave32) per CDNA5 ISA layouts.
// A-matrix 16x32 (MxK): lane L -> row m=L&15, half=L>>4.
//   elem j in [0,8):  k = j + 8*half        (16B contiguous)
//   elem j in [8,16): k = j + 8 + 8*half    (16B contiguous at +16 elems)
// B-matrix 32x16 (KxN): lane L -> col n=L&15, half=L>>4.
//   elem j: same k mapping as A (two 16B chunks at +0 and +16 elems)
// ---------------------------------------------------------------------------
__device__ __forceinline__ v16h load_a_f16(const _Float16* base, int row0, int ld,
                                           int k0, int lane) {
  int m = lane & 15, half = (lane >> 4) & 1;
  const _Float16* p = base + (size_t)(row0 + m) * ld + k0 + half * 8;
  v16h r;
  uint4 q0 = *(const uint4*)(p);
  uint4 q1 = *(const uint4*)(p + 16);
  __builtin_memcpy((char*)&r,      &q0, 16);
  __builtin_memcpy((char*)&r + 16, &q1, 16);
  return r;
}

__device__ __forceinline__ v16h load_b_f16(const _Float16* base, int n0, int ld,
                                           int k0, int lane) {
  int n = lane & 15, half = (lane >> 4) & 1;
  const _Float16* p = base + (size_t)(n0 + n) * ld + k0 + half * 8;
  v16h r;
  uint4 q0 = *(const uint4*)(p);
  uint4 q1 = *(const uint4*)(p + 16);
  __builtin_memcpy((char*)&r,      &q0, 16);
  __builtin_memcpy((char*)&r + 16, &q1, 16);
  return r;
}

__device__ __forceinline__ v8f vzero8() {
  v8f z = {0.f, 0.f, 0.f, 0.f, 0.f, 0.f, 0.f, 0.f};
  return z;
}

__device__ __forceinline__ float sigmoidf_(float x) {
  return 1.0f / (1.0f + __expf(-x));
}

// ---------------------------------------------------------------------------
// Problem constants
// ---------------------------------------------------------------------------
#define BB 64
#define TT 512
#define EE 128
#define HH 256
#define G4 1024   // 4*H

// ---------------------------------------------------------------------------
// Kernel 0: zero the sig_out accumulator region of d_out
// ---------------------------------------------------------------------------
__global__ void k_zero_out(float* out) {
  if (threadIdx.x < BB) out[threadIdx.x] = 0.0f;
}

// ---------------------------------------------------------------------------
// Kernel 1: fp32 -> fp16 weight conversion
// ---------------------------------------------------------------------------
__global__ void k_cvt_f16(const float* __restrict__ src, _Float16* __restrict__ dst, int n) {
  int i = blockIdx.x * blockDim.x + threadIdx.x;
  if (i < n) dst[i] = (_Float16)src[i];
}

// ---------------------------------------------------------------------------
// Kernel 2: embedding gather -> fp16 A matrix in [t*B + b, E] row order
// ---------------------------------------------------------------------------
__global__ void k_embed(const int* __restrict__ x, const float* __restrict__ emb,
                        _Float16* __restrict__ A) {
  int idx = blockIdx.x * blockDim.x + threadIdx.x;   // over B*T*E
  if (idx >= BB * TT * EE) return;
  int e = idx & (EE - 1);
  int r = idx >> 7;            // r = t*B + b
  int b = r & (BB - 1);
  int t = r >> 6;
  int tok = x[b * TT + t];
  A[idx] = (_Float16)emb[(size_t)tok * EE + e];
}

// ---------------------------------------------------------------------------
// Kernel 3: gx[r, g] = A_emb[r,:] . W_ih[g,:] + b_ih[g]   (M=32768,N=1024,K=128)
// one 16x16 tile per wave, 8 waves per block
// ---------------------------------------------------------------------------
__global__ __launch_bounds__(256) void k_gx(const _Float16* __restrict__ A,
                                            const _Float16* __restrict__ Wih,
                                            const float* __restrict__ b_ih,
                                            float* __restrict__ gx) {
  int lane = threadIdx.x & 31, wv = threadIdx.x >> 5;
  int tile = blockIdx.x * 8 + wv;        // 131072 tiles total
  int mt = tile >> 6, nt = tile & 63;
  int r0 = mt * 16, n0 = nt * 16;
  int half = lane >> 4, ln = lane & 15;

  v8f acc = vzero8();
#pragma unroll
  for (int kc = 0; kc < 4; ++kc) {
    v16h a = load_a_f16(A,   r0, EE, kc * 32, lane);
    v16h b = load_b_f16(Wih, n0, EE, kc * 32, lane);
    acc = __builtin_amdgcn_wmma_f32_16x16x32_f16(false, a, false, b,
                                                 (short)0, acc, false, false);
  }
  float bias = b_ih[n0 + ln];
#pragma unroll
  for (int v = 0; v < 8; ++v) {
    int m = v + 8 * half;
    gx[(size_t)(r0 + m) * G4 + n0 + ln] = acc[v] + bias;
  }
}

// ---------------------------------------------------------------------------
// Kernel 4: sequential LSTM scan. One persistent workgroup (32 waves).
// h double-buffered in LDS (fp16), c lives in registers (each (b,hh) owned by
// exactly one lane). One barrier per timestep.
// ---------------------------------------------------------------------------
__global__ __launch_bounds__(1024) void k_scan(const float* __restrict__ gx,
                                               const _Float16* __restrict__ Whh,
                                               const float* __restrict__ b_hh,
                                               _Float16* __restrict__ hs,
                                               float* __restrict__ out) {
  __shared__ _Float16 hbuf[2][BB * HH];   // 2 x 32 KB = 64 KB
  int tid = threadIdx.x, lane = tid & 31, wv = tid >> 5;
  int half = lane >> 4, ln = lane & 15;

  for (int i = tid; i < 2 * BB * HH; i += 1024)
    ((_Float16*)hbuf)[i] = (_Float16)0.0f;
  __syncthreads();

  float c_reg[2][8];
#pragma unroll
  for (int q = 0; q < 2; ++q)
#pragma unroll
    for (int v = 0; v < 8; ++v) c_reg[q][v] = 0.0f;

  int cur = 0;
  for (int t = 0; t < TT; ++t) {
    const _Float16* hc = hbuf[cur];
    _Float16*       hn = hbuf[cur ^ 1];
    const float* gxt = gx + (size_t)t * BB * G4;

#pragma unroll
    for (int q = 0; q < 2; ++q) {
      int combo = wv + q * 32;            // 64 combos: (batch tile mt, h tile j)
      int mt = combo >> 4, j = combo & 15;

      v8f acc0 = vzero8(), acc1 = vzero8(), acc2 = vzero8(), acc3 = vzero8();
#pragma unroll
      for (int kc = 0; kc < 8; ++kc) {
        v16h a = load_a_f16(hc, mt * 16, HH, kc * 32, lane);
        v16h b0 = load_b_f16(Whh, 0 * HH + j * 16, HH, kc * 32, lane);
        acc0 = __builtin_amdgcn_wmma_f32_16x16x32_f16(false, a, false, b0, (short)0, acc0, false, false);
        v16h b1 = load_b_f16(Whh, 1 * HH + j * 16, HH, kc * 32, lane);
        acc1 = __builtin_amdgcn_wmma_f32_16x16x32_f16(false, a, false, b1, (short)0, acc1, false, false);
        v16h b2 = load_b_f16(Whh, 2 * HH + j * 16, HH, kc * 32, lane);
        acc2 = __builtin_amdgcn_wmma_f32_16x16x32_f16(false, a, false, b2, (short)0, acc2, false, false);
        v16h b3 = load_b_f16(Whh, 3 * HH + j * 16, HH, kc * 32, lane);
        acc3 = __builtin_amdgcn_wmma_f32_16x16x32_f16(false, a, false, b3, (short)0, acc3, false, false);
      }

      int hh = j * 16 + ln;
      float bh0 = b_hh[0 * HH + hh], bh1 = b_hh[1 * HH + hh];
      float bh2 = b_hh[2 * HH + hh], bh3 = b_hh[3 * HH + hh];
#pragma unroll
      for (int v = 0; v < 8; ++v) {
        int bb = mt * 16 + v + 8 * half;
        const float* grow = gxt + (size_t)bb * G4;
        float gi = acc0[v] + grow[0 * HH + hh] + bh0;
        float gf = acc1[v] + grow[1 * HH + hh] + bh1;
        float gg = acc2[v] + grow[2 * HH + hh] + bh2;
        float go = acc3[v] + grow[3 * HH + hh] + bh3;
        float I = sigmoidf_(gi), F = sigmoidf_(gf);
        float G = tanhf(gg),     O = sigmoidf_(go);
        float cn = F * c_reg[q][v] + I * G;
        c_reg[q][v] = cn;
        float hv = O * tanhf(cn);
        hn[bb * HH + hh] = (_Float16)hv;
        hs[((size_t)(t * BB + bb)) * HH + hh] = (_Float16)hv;
        if (t == TT - 1) {
          out[BB + bb * HH + hh]           = hv;   // h_n
          out[BB + BB * HH + bb * HH + hh] = cn;   // c_n
        }
      }
    }
    __syncthreads();
    cur ^= 1;
  }
}

// ---------------------------------------------------------------------------
// Kernel 5: head. out_m = hs @ Wm^T + bm (WMMA), fused with Wf dot + sigmoid +
// mean over T. One 16-row block per workgroup (8 waves x 2 n-tiles).
// ---------------------------------------------------------------------------
__global__ __launch_bounds__(256) void k_head(const _Float16* __restrict__ hs,
                                              const _Float16* __restrict__ Wm16,
                                              const float* __restrict__ bm,
                                              const _Float16* __restrict__ Wf16,
                                              const float* __restrict__ bf,
                                              float* __restrict__ out) {
  __shared__ float partial[16];
  int tid = threadIdx.x, lane = tid & 31, wv = tid >> 5;
  int half = lane >> 4, ln = lane & 15;
  int r0 = blockIdx.x * 16;

  if (tid < 16) partial[tid] = 0.0f;
  __syncthreads();

  float pm[8];
#pragma unroll
  for (int v = 0; v < 8; ++v) pm[v] = 0.0f;

#pragma unroll
  for (int q = 0; q < 2; ++q) {
    int n0 = (wv * 2 + q) * 16;
    int ng = n0 + ln;
    v8f acc = vzero8();
#pragma unroll
    for (int kc = 0; kc < 8; ++kc) {
      v16h a = load_a_f16(hs,   r0, HH, kc * 32, lane);
      v16h b = load_b_f16(Wm16, n0, HH, kc * 32, lane);
      acc = __builtin_amdgcn_wmma_f32_16x16x32_f16(false, a, false, b,
                                                   (short)0, acc, false, false);
    }
    float bmv = bm[ng];
    float wfv = (float)Wf16[ng];
#pragma unroll
    for (int v = 0; v < 8; ++v) pm[v] += (acc[v] + bmv) * wfv;
  }

  // reduce over the 16 lanes sharing the same m (xor offsets stay in-half)
#pragma unroll
  for (int off = 8; off > 0; off >>= 1)
#pragma unroll
    for (int v = 0; v < 8; ++v) pm[v] += __shfl_xor(pm[v], off, 32);

  if (ln == 0) {
#pragma unroll
    for (int v = 0; v < 8; ++v) atomicAdd(&partial[v + 8 * half], pm[v]);
  }
  __syncthreads();

  if (tid < 16) {
    float s = sigmoidf_(partial[tid] + bf[0]);
    int r = r0 + tid;           // r = t*B + b
    int b = r & (BB - 1);
    atomicAdd(out + b, s * (1.0f / (float)TT));
  }
}

// ---------------------------------------------------------------------------
// Host launch
// ---------------------------------------------------------------------------
extern "C" void kernel_launch(void* const* d_in, const int* in_sizes, int n_in,
                              void* d_out, int out_size, void* d_ws, size_t ws_size,
                              hipStream_t stream) {
  (void)in_sizes; (void)n_in; (void)out_size; (void)ws_size;
  const int*   x     = (const int*)  d_in[0];
  const float* emb   = (const float*)d_in[1];
  const float* W_ih  = (const float*)d_in[2];
  const float* W_hh  = (const float*)d_in[3];
  const float* b_ih  = (const float*)d_in[4];
  const float* b_hh  = (const float*)d_in[5];
  const float* Wm    = (const float*)d_in[6];
  const float* bm    = (const float*)d_in[7];
  const float* Wf    = (const float*)d_in[8];
  const float* bf    = (const float*)d_in[9];
  float* out = (float*)d_out;

  char* ws = (char*)d_ws;
  // workspace layout (bytes)
  _Float16* A_emb  = (_Float16*)(ws + 0);                    //  8 MB
  _Float16* W_ih16 = (_Float16*)(ws + (8ULL  << 20));        // 256 KB
  _Float16* W_hh16 = (_Float16*)(ws + (8ULL  << 20) + 262144ULL);
  _Float16* Wm16   = (_Float16*)(ws + (8ULL  << 20) + 262144ULL + 524288ULL);
  _Float16* Wf16   = (_Float16*)(ws + (8ULL  << 20) + 262144ULL + 524288ULL + 131072ULL);
  float*    gxbuf  = (float*)   (ws + (16ULL << 20));        // 128 MB
  _Float16* hsbuf  = (_Float16*)(ws + (16ULL << 20) + (128ULL << 20));  // 16 MB

  k_zero_out<<<1, 64, 0, stream>>>(out);

  k_cvt_f16<<<(G4 * EE + 255) / 256, 256, 0, stream>>>(W_ih, W_ih16, G4 * EE);
  k_cvt_f16<<<(G4 * HH + 255) / 256, 256, 0, stream>>>(W_hh, W_hh16, G4 * HH);
  k_cvt_f16<<<(HH * HH + 255) / 256, 256, 0, stream>>>(Wm,   Wm16,   HH * HH);
  k_cvt_f16<<<1, 256, 0, stream>>>(Wf, Wf16, HH);

  k_embed<<<(BB * TT * EE + 255) / 256, 256, 0, stream>>>(x, emb, A_emb);

  // 32768/16 * 1024/16 = 131072 tiles, 8 waves per 256-thread block
  k_gx<<<131072 / 8, 256, 0, stream>>>(A_emb, W_ih16, b_ih, gxbuf);

  k_scan<<<1, 1024, 0, stream>>>(gxbuf, W_hh16, b_hh, hsbuf, out);

  k_head<<<(BB * TT) / 16, 256, 0, stream>>>(hsbuf, Wm16, bm, Wf16, bf, out);
}
